// SNN_LSTM_Model_58729382806127
// MI455X (gfx1250) — matrix operand order repaired and checked by reading, not compiled, not run on hardware
//
#include <hip/hip_runtime.h>
#include <hip/hip_bf16.h>

// ---------------------------------------------------------------------------
// SNN-LSTM (B=32, T=512, D=256, H=512, O=128) for gfx1250.
// Persistent fused LSTM+LIF scan using v_wmma_f32_16x16x32_bf16.
// 16 workgroups; WG j owns hidden slice [32j,32j+32) across all 4 gates so
// the per-step elementwise update is WG-local; one grid barrier per step.
// Weights pre-swizzled into WMMA operand layout, L2-resident (192MB L2).
// ---------------------------------------------------------------------------

typedef __attribute__((ext_vector_type(16))) __bf16 v16bf;
typedef __attribute__((ext_vector_type(8)))  __bf16 v8bf;
typedef __attribute__((ext_vector_type(8)))  float  v8f;

#define B_    32
#define T_    512
#define D_    256
#define H_    512
#define O_    128
#define G4    2048          // 4*H
#define NWG   16
#define KT_REC 24           // (D_+H_)/32 k-steps for recurrent GEMM
#define KT_FC  16           // H_/32 for final FC

// workspace layout (bytes)
#define OFF_CNT   0
#define OFF_XBF   256
#define SZ_XBF    (T_*B_*D_*2)                 // 8,388,608  x in [T][B][D] bf16
#define OFF_BSWZ  (OFF_XBF + SZ_XBF)
#define SZ_BSWZ   (128*KT_REC*32*16*2)         // 3,145,728  [Wih^T;Whh^T] swizzled
#define OFF_BIAS  (OFF_BSWZ + SZ_BSWZ)
#define SZ_BIAS   (G4*4)                       // b_ih+b_hh f32
#define OFF_HBF   (OFF_BIAS + SZ_BIAS)
#define SZ_HBF    (2*B_*H_*2)                  // double-buffered h, bf16
#define OFF_SPK   (OFF_HBF + SZ_HBF)
#define SZ_SPK    (B_*H_*4)                    // last-step spikes f32
#define OFF_WFC   (OFF_SPK + SZ_SPK)
#define SZ_WFC    (8*KT_FC*32*16*2)            // W_fc swizzled

__device__ __forceinline__ v8f wmma_bf16(v16bf a, v16bf b, v8f c) {
  return __builtin_amdgcn_wmma_f32_16x16x32_bf16(
      /*neg_a=*/false, a, /*neg_b=*/false, b,
      /*c_mod=*/(short)0, c, /*reuse_a=*/false, /*reuse_b=*/false);
}

// 16-bit A-matrix 16x32 lane layout: lane<16 holds K {0..7,16..23} of row M=lane,
// lane>=16 holds K {8..15,24..31} of row M=lane-16 (ISA 7.12.2).
__device__ __forceinline__ v16bf load_a16(const __bf16* rowp, bool lane_hi) {
  const __bf16* p0 = rowp + (lane_hi ? 8  : 0);
  const __bf16* p1 = rowp + (lane_hi ? 24 : 16);
  v8bf lo = *(const v8bf*)p0;
  v8bf hi = *(const v8bf*)p1;
  return __builtin_shufflevector(lo, hi, 0,1,2,3,4,5,6,7,8,9,10,11,12,13,14,15);
}

__device__ __forceinline__ float fsig(float x)  { return 1.0f / (1.0f + __expf(-x)); }
__device__ __forceinline__ float ftanh(float x) { float e = __expf(-2.0f*x); return (1.0f - e)/(1.0f + e); }

// --------------------------- setup kernels ---------------------------------

__global__ void init_k(unsigned* cnt, unsigned* hbf_u32) {
  int i = blockIdx.x*blockDim.x + threadIdx.x;
  if (i == 0) *cnt = 0u;
  for (int j = i; j < (SZ_HBF/4); j += gridDim.x*blockDim.x) hbf_u32[j] = 0u;
}

// x[B][T][D] f32 -> xbf[T][B][D] bf16
__global__ void xcvt_k(const float* __restrict__ x, __bf16* __restrict__ xbf) {
  int i = blockIdx.x*256 + threadIdx.x;          // T_*B_*D_ = 4194304 exactly
  int t = i >> 13;                                // /(B*D)
  int r = i & 8191;
  int b = r >> 8;
  int d = r & 255;
  xbf[i] = (__bf16)x[((size_t)b*T_ + t)*D_ + d];
}

// Build B-operand for recurrent GEMM: columns permuted so WG j owns
// {gate*512 + 32j + hl}. B lane layout: lane<16 -> K=kt*32+{0..7,16..23}, lane>=16 -> K=kt*32+{8..15,24..31}.
__global__ void bswz_k(const float* __restrict__ wih, const float* __restrict__ whh,
                       __bf16* __restrict__ bswz) {
  int i = blockIdx.x*256 + threadIdx.x;           // 128*24*32*16 = 1572864 exactly
  int e    = i & 15;
  int lane = (i >> 4) & 31;
  int kt   = (i >> 9) % KT_REC;
  int nt   = i / (KT_REC * 512);
  int c      = nt*16 + (lane & 15);               // permuted column id
  int wg     = c >> 7;
  int within = c & 127;
  int gate   = within >> 5;
  int hl     = within & 31;
  int Grow   = gate*H_ + wg*32 + hl;              // row of W_ih / W_hh
  int k      = kt*32 + ((e < 8) ? e : 8 + e) + ((lane < 16) ? 0 : 8);
  float v = (k < D_) ? wih[(size_t)Grow*D_ + k] : whh[(size_t)Grow*H_ + (k - D_)];
  bswz[i] = (__bf16)v;
}

__global__ void bias_k(const float* bih, const float* bhh, float* bias) {
  int i = blockIdx.x*256 + threadIdx.x;
  if (i < G4) bias[i] = bih[i] + bhh[i];
}

__global__ void wfc_k(const float* __restrict__ wfc, __bf16* __restrict__ out) {
  int i = blockIdx.x*256 + threadIdx.x;           // 8*16*32*16 = 65536 exactly
  int e    = i & 15;
  int lane = (i >> 4) & 31;
  int kt   = (i >> 9) % KT_FC;
  int nt   = i / (KT_FC * 512);
  int n = nt*16 + (lane & 15);
  int k = kt*32 + ((e < 8) ? e : 8 + e) + ((lane < 16) ? 0 : 8);
  out[i] = (__bf16)wfc[(size_t)n*H_ + k];
}

// ----------------------- persistent LSTM+LIF scan --------------------------

__global__ void __launch_bounds__(256, 1)
snn_k(const __bf16* __restrict__ xbf, const __bf16* __restrict__ bswz,
      const float* __restrict__ bias, __bf16* __restrict__ hbf,
      float* __restrict__ spk, unsigned* __restrict__ cnt)
{
  __shared__ float accs[32][128];     // gate pre-activations, [batch][4*32 local cols]
  __shared__ float bias_s[128];

  const int  tid  = threadIdx.x;
  const int  lane = tid & 31;
  const int  wave = tid >> 5;         // 8 waves; wave w owns n-tile w (16 cols)
  const int  wg   = blockIdx.x;       // 16 WGs; WG owns hidden [32*wg, 32*wg+32)
  const bool lhi  = lane >= 16;
  const int  mrow = lane & 15;

  if (tid < 128) {
    int gate = tid >> 5, hl = tid & 31;
    bias_s[tid] = bias[gate*H_ + wg*32 + hl];
  }
  __syncthreads();

  const int nt = wg*8 + wave;
  const __bf16* bbase = bswz + (size_t)nt * KT_REC * 512;   // 512 bf16 per (nt,kt) block

  float cst[4], mem[4];
  #pragma unroll
  for (int q = 0; q < 4; ++q) { cst[q] = 0.0f; mem[q] = 0.0f; }

  for (int t = 0; t < T_; ++t) {
    // acquire: invalidate caches so this WGP sees other WGPs' h stores from t-1
    __threadfence();

    const int cur = t & 1, nxt = cur ^ 1;
    const __bf16* xrow = xbf + (size_t)t * B_ * D_;
    const __bf16* hrow = hbf + (size_t)cur * B_ * H_;

    v8f acc0 = {}; v8f acc1 = {};

    // ---- x phase: K = 0..255, fixed base pointers -> immediate offsets ----
    {
      const __bf16* a0b = xrow + (size_t)mrow        * D_;
      const __bf16* a1b = xrow + (size_t)(mrow + 16) * D_;
      #pragma unroll
      for (int kt = 0; kt < 8; ++kt) {
        v16bf a0 = load_a16(a0b + kt*32, lhi);
        v16bf a1 = load_a16(a1b + kt*32, lhi);
        v16bf bv = *(const v16bf*)(bbase + (size_t)kt*512 + lane*16);  // 32B/lane
        acc0 = wmma_bf16(a0, bv, acc0);
        acc1 = wmma_bf16(a1, bv, acc1);
      }
    }
    // ---- h phase: K = 256..767 ----
    {
      const __bf16* a0b = hrow + (size_t)mrow        * H_;
      const __bf16* a1b = hrow + (size_t)(mrow + 16) * H_;
      const __bf16* bb2 = bbase + (size_t)8*512;
      #pragma unroll 8
      for (int kt = 0; kt < 16; ++kt) {
        v16bf a0 = load_a16(a0b + kt*32, lhi);
        v16bf a1 = load_a16(a1b + kt*32, lhi);
        v16bf bv = *(const v16bf*)(bb2 + (size_t)kt*512 + lane*16);
        acc0 = wmma_bf16(a0, bv, acc0);
        acc1 = wmma_bf16(a1, bv, acc1);
      }
    }

    // prefetch next timestep's x rows while we do elementwise + barrier
    if (t + 1 < T_) {
      const __bf16* xn = xbf + (size_t)(t + 1) * B_ * D_ + (size_t)mrow * D_;
      __builtin_prefetch(xn, 0, 1);
      __builtin_prefetch(xn + (size_t)16 * D_, 0, 1);
    }

    // stage accumulators: D layout VGPR r, lane l -> row r+8*(l>=16), col l&15
    const int colw = wave*16 + mrow;
    const int rb   = lhi ? 8 : 0;
    #pragma unroll
    for (int r = 0; r < 8; ++r) accs[rb + r][colw]      = acc0[r];
    #pragma unroll
    for (int r = 0; r < 8; ++r) accs[16 + rb + r][colw] = acc1[r];
    __syncthreads();

    // LSTM cell + LIF for this WG's 32 hidden x 32 batch (4 elems/thread)
    #pragma unroll
    for (int q = 0; q < 4; ++q) {
      int p  = tid + 256*q;
      int b  = p >> 5;
      int hl = p & 31;
      float iv = fsig (accs[b][hl]      + bias_s[hl]);
      float fv = fsig (accs[b][32 + hl] + bias_s[32 + hl]);
      float gv = ftanh(accs[b][64 + hl] + bias_s[64 + hl]);
      float ov = fsig (accs[b][96 + hl] + bias_s[96 + hl]);
      float cv = fv * cst[q] + iv * gv;
      cst[q] = cv;
      float ht = ov * ftanh(cv);
      float mv = 0.9f * mem[q] + ht;             // BETA = 0.9
      float sp = (mv > 1.0f) ? 1.0f : 0.0f;      // THRESH = 1.0, strict >
      mem[q] = mv - sp;                           // reset by subtraction
      hbf[(size_t)nxt*B_*H_ + (size_t)b*H_ + wg*32 + hl] = (__bf16)ht;
      if (t == T_ - 1) spk[(size_t)b*H_ + wg*32 + hl] = sp;
    }

    // release h stores, then one grid barrier per timestep (sleep-backoff spin)
    __threadfence();
    __syncthreads();
    if (tid == 0) {
      __hip_atomic_fetch_add(cnt, 1u, __ATOMIC_ACQ_REL, __HIP_MEMORY_SCOPE_AGENT);
      const unsigned target = (unsigned)NWG * (unsigned)(t + 1);
      while (__hip_atomic_load(cnt, __ATOMIC_ACQUIRE, __HIP_MEMORY_SCOPE_AGENT) < target)
        __builtin_amdgcn_s_sleep(2);
    }
    __syncthreads();
  }
}

// ----------------------------- final FC ------------------------------------

__global__ void __launch_bounds__(256, 1)
fc_k(const float* __restrict__ spk, const __bf16* __restrict__ wfcs,
     const float* __restrict__ bfc, float* __restrict__ out)
{
  const int  tid  = threadIdx.x;
  const int  lane = tid & 31;
  const int  wave = tid >> 5;
  const bool lhi  = lane >= 16;
  const int  mrow = lane & 15;
  const int  o1 = lhi ? 8 : 0, o2 = lhi ? 24 : 16;

  v8f acc0 = {}; v8f acc1 = {};
  const __bf16* bb = wfcs + (size_t)wave * KT_FC * 512;
  for (int kt = 0; kt < KT_FC; ++kt) {
    const float* r0 = spk + (size_t)mrow        * H_ + kt*32;
    const float* r1 = spk + (size_t)(mrow + 16) * H_ + kt*32;
    v16bf a0, a1;
    #pragma unroll
    for (int e = 0; e < 8; ++e) {      // spikes are exactly 0/1 -> bf16 exact
      a0[e]   = (__bf16)r0[o1 + e];
      a0[8+e] = (__bf16)r0[o2 + e];
      a1[e]   = (__bf16)r1[o1 + e];
      a1[8+e] = (__bf16)r1[o2 + e];
    }
    v16bf bv = *(const v16bf*)(bb + (size_t)kt*512 + lane*16);
    acc0 = wmma_bf16(a0, bv, acc0);
    acc1 = wmma_bf16(a1, bv, acc1);
  }
  const int ocol = wave*16 + mrow;
  const int rb   = lhi ? 8 : 0;
  const float bo = bfc[ocol];
  #pragma unroll
  for (int r = 0; r < 8; ++r) {
    out[(size_t)(rb + r)*O_ + ocol]      = acc0[r] + bo;
    out[(size_t)(16 + rb + r)*O_ + ocol] = acc1[r] + bo;
  }
}

// ------------------------------- launch ------------------------------------

extern "C" void kernel_launch(void* const* d_in, const int* in_sizes, int n_in,
                              void* d_out, int out_size, void* d_ws, size_t ws_size,
                              hipStream_t stream) {
  (void)in_sizes; (void)n_in; (void)out_size; (void)ws_size;
  const float* x   = (const float*)d_in[0];
  const float* wih = (const float*)d_in[1];
  const float* whh = (const float*)d_in[2];
  const float* bih = (const float*)d_in[3];
  const float* bhh = (const float*)d_in[4];
  const float* wfc = (const float*)d_in[5];
  const float* bfc = (const float*)d_in[6];

  char* ws = (char*)d_ws;
  unsigned* cnt  = (unsigned*)(ws + OFF_CNT);
  __bf16*   xbf  = (__bf16*)(ws + OFF_XBF);
  __bf16*   bswz = (__bf16*)(ws + OFF_BSWZ);
  float*    bias = (float*)(ws + OFF_BIAS);
  __bf16*   hbf  = (__bf16*)(ws + OFF_HBF);
  float*    spk  = (float*)(ws + OFF_SPK);
  __bf16*   wfcs = (__bf16*)(ws + OFF_WFC);

  init_k<<<64,   256, 0, stream>>>(cnt, (unsigned*)hbf);
  xcvt_k<<<16384,256, 0, stream>>>(x, xbf);
  bswz_k<<<6144, 256, 0, stream>>>(wih, whh, bswz);
  bias_k<<<8,    256, 0, stream>>>(bih, bhh, bias);
  wfc_k <<<256,  256, 0, stream>>>(wfc, wfcs);
  snn_k <<<NWG,  256, 0, stream>>>(xbf, bswz, bias, hbf, spk, cnt);
  fc_k  <<<1,    256, 0, stream>>>(spk, wfcs, bfc, (float*)d_out);
}
